// AdvancedHeteroLinkPredictor_62577673503221
// MI455X (gfx1250) — matrix operand order repaired and checked by reading, not compiled, not run on hardware
//
#include <hip/hip_runtime.h>
#include <hip/hip_bf16.h>

// ---------------------------------------------------------------------------
// Types
// ---------------------------------------------------------------------------
typedef _Float16 h16;
typedef __attribute__((ext_vector_type(16))) _Float16 v16h;
typedef __attribute__((ext_vector_type(8)))  _Float16 v8h;
typedef __attribute__((ext_vector_type(4)))  _Float16 v4h;
typedef __attribute__((ext_vector_type(8)))  float    v8f;

#define NG_  20000
#define ND_  20000
#define DD_  256
#define EM_  262144
#define EP_  131072
#define EPS_ 1e-5f

// ---------------------------------------------------------------------------
// WMMA GEMM: C[M,N] = epilogue(A[M,K](f16) @ B[K,N](f16))
// Per-wave 16x32 tile (2 WMMAs share one A fragment), 8 waves/block
// arranged 4(M) x 2(N) -> 64x64 block tile, K step 32.
// ---------------------------------------------------------------------------
template<bool BIAS, bool RELU, bool AUX, bool ACCUM>
__global__ __launch_bounds__(256)
void wmma_gemm(const h16* __restrict__ A, const h16* __restrict__ B,
               const float* __restrict__ bias, const float* __restrict__ aux,
               float* __restrict__ C, h16* __restrict__ C16,
               int M, int N, int K)
{
    const int lane = threadIdx.x & 31;
    const int wave = threadIdx.x >> 5;
    const int wm   = wave & 3;            // 4 waves along M
    const int wn   = wave >> 2;           // 2 waves along N (32 cols each)
    const int row0 = blockIdx.y * 64 + wm * 16;
    const int col0 = blockIdx.x * 64 + wn * 32;
    const int half = lane >> 4;           // K-half selector for A frag
    const int r    = lane & 15;

    // A row for this lane (clamped so EXEC stays all-ones for WMMA)
    int am = row0 + r; if (am > M - 1) am = M - 1;
    const h16* __restrict__ Arow = A + (size_t)am * K;

    v8f acc0 = {}, acc1 = {};
    for (int kb = 0; kb < K; kb += 32) {
        // prefetch next k-step of the A stream (global_prefetch_b8; speculative-safe)
        __builtin_prefetch((const void*)(Arow + kb + 32), 0, 1);

        // A fragment: elements 0..7 -> K = kb+half*8+0..7 ; 8..15 -> +16
        v8h alo = *(const v8h*)(Arow + kb + half * 8);
        v8h ahi = *(const v8h*)(Arow + kb + 16 + half * 8);
        v16h a;
#pragma unroll
        for (int i = 0; i < 8; ++i) { a[i] = alo[i]; a[i + 8] = ahi[i]; }

        // B fragments: lane holds column N = col0+r (+16 for b1); elements 0..7 -> K = kb+half*8+0..7 ; 8..15 -> +16
        const h16* __restrict__ Bcol = B + col0 + r;
        v16h b0, b1;
#pragma unroll
        for (int i = 0; i < 8; ++i) {
            size_t klo = (size_t)(kb + half * 8 + i) * N;
            size_t khi = (size_t)(kb + 16 + half * 8 + i) * N;
            b0[i] = Bcol[klo];      b0[i + 8] = Bcol[khi];
            b1[i] = Bcol[klo + 16]; b1[i + 8] = Bcol[khi + 16];
        }

        acc0 = __builtin_amdgcn_wmma_f32_16x16x32_f16(
            false, a, false, b0, (short)0, acc0, false, false);
        acc1 = __builtin_amdgcn_wmma_f32_16x16x32_f16(
            false, a, false, b1, (short)0, acc1, false, false);
    }

    const int colA = col0 + r;
#pragma unroll
    for (int v = 0; v < 8; ++v) {
        int row = row0 + v + half * 8;
        if (row < M) {
            size_t rowoff = (size_t)row * N;
#pragma unroll
            for (int t = 0; t < 2; ++t) {
                int   col = colA + t * 16;
                float val = t ? acc1[v] : acc0[v];
                if (BIAS) val += bias[col];
                if (AUX)  val += aux[rowoff + col];
                if (RELU) val = fmaxf(val, 0.f);
                size_t idx = rowoff + col;
                if (ACCUM) { if (C) C[idx] += val; }
                else       { if (C) C[idx]  = val; }
                if (C16) C16[idx] = (h16)val;
            }
        }
    }
}

// ---------------------------------------------------------------------------
// Elementwise / graph kernels
// ---------------------------------------------------------------------------
__global__ void fill_f32(float* p, float v, size_t n) {
    size_t id = (size_t)blockIdx.x * blockDim.x + threadIdx.x;
    if (id < n) p[id] = v;
}

__global__ void cvt_f32_f16(const float* __restrict__ in, h16* __restrict__ out, size_t n) {
    size_t id = (size_t)blockIdx.x * blockDim.x + threadIdx.x;
    if (id < n) out[id] = (h16)in[id];
}

// one thread = (edge, 4 consecutive floats): gather src row, atomic-add dst row
__global__ void scatter_rows(const float* __restrict__ h, const int* __restrict__ src,
                             const int* __restrict__ dst, float* __restrict__ acc,
                             int nedges)
{
    size_t id = (size_t)blockIdx.x * blockDim.x + threadIdx.x;
    if (id >= (size_t)nedges * 64) return;
    int e = (int)(id >> 6);
    int c = (int)(id & 63) << 2;
    int s = src[e], d = dst[e];
    float4 v = *(const float4*)(h + (size_t)s * DD_ + c);
    float* p = acc + (size_t)d * DD_ + c;
    atomicAdd(p + 0, v.x); atomicAdd(p + 1, v.y);
    atomicAdd(p + 2, v.z); atomicAdd(p + 3, v.w);
}

__global__ void scatter_cnt(const int* __restrict__ dst, float* __restrict__ cnt, int n) {
    int e = blockIdx.x * blockDim.x + threadIdx.x;
    if (e < n) atomicAdd(cnt + dst[e], 1.f);
}

__global__ void div_cvt(const float* __restrict__ acc, const float* __restrict__ cnt,
                        h16* __restrict__ out, size_t total)
{
    size_t id = (size_t)blockIdx.x * blockDim.x + threadIdx.x;
    if (id >= total) return;
    int row = (int)(id >> 8);
    out[id] = (h16)(acc[id] / fmaxf(cnt[row], 1.f));
}

// LayerNorm over D=256 with optional residual; writes f32 + f16 shadow
__global__ __launch_bounds__(256)
void ln_res(const float* __restrict__ x, const float* __restrict__ res,
            const float* __restrict__ g, const float* __restrict__ b,
            float* __restrict__ out, h16* __restrict__ out16)
{
    __shared__ float sh[256];
    int row = blockIdx.x, t = threadIdx.x;
    size_t idx = (size_t)row * DD_ + t;
    float v = x[idx];
    if (res) v += res[idx];
    sh[t] = v; __syncthreads();
    for (int s = 128; s > 0; s >>= 1) { if (t < s) sh[t] += sh[t + s]; __syncthreads(); }
    float mean = sh[0] * (1.f / DD_); __syncthreads();
    float d = v - mean; sh[t] = d * d; __syncthreads();
    for (int s = 128; s > 0; s >>= 1) { if (t < s) sh[t] += sh[t + s]; __syncthreads(); }
    float var = sh[0] * (1.f / DD_);
    float o = d * rsqrtf(var + EPS_) * g[t] + b[t];
    out[idx] = o; out16[idx] = (h16)o;
}

// ef[e] = concat(hg16[src[e]], hd16[dst[e]])  (thread = 4 halves)
__global__ void gather_concat(const h16* __restrict__ hg, const h16* __restrict__ hd,
                              const int* __restrict__ src, const int* __restrict__ dst,
                              h16* __restrict__ ef, int E)
{
    size_t id = (size_t)blockIdx.x * blockDim.x + threadIdx.x;
    if (id >= (size_t)E * 128) return;
    int e = (int)(id >> 7);
    int j = (int)(id & 127) << 2;
    const h16* p = (j < DD_) ? (hg + (size_t)src[e] * DD_ + j)
                             : (hd + (size_t)dst[e] * DD_ + (j - DD_));
    *(v4h*)(ef + (size_t)e * (2 * DD_) + j) = *(const v4h*)p;
}

// column-wise sums for BatchNorm (thread = column, block = 256-row chunk)
__global__ __launch_bounds__(256)
void bn_stats(const h16* __restrict__ t, float* __restrict__ sum,
              float* __restrict__ sumsq, int rows_per_block)
{
    int col = threadIdx.x;
    size_t base = (size_t)blockIdx.x * rows_per_block;
    float s = 0.f, s2 = 0.f;
    for (int r = 0; r < rows_per_block; ++r) {
        float v = (float)t[(base + r) * DD_ + col];
        s += v; s2 += v * v;
    }
    atomicAdd(sum + col, s);
    atomicAdd(sumsq + col, s2);
}

__global__ void bn_apply(const h16* __restrict__ t, const float* __restrict__ sum,
                         const float* __restrict__ sumsq, const float* __restrict__ g,
                         const float* __restrict__ b, const float* __restrict__ emb,
                         h16* __restrict__ out, size_t total, float invE)
{
    size_t id = (size_t)blockIdx.x * blockDim.x + threadIdx.x;
    if (id >= total) return;
    int col = (int)(id & 255);
    float mu  = sum[col] * invE;
    float var = sumsq[col] * invE - mu * mu;
    float v = (float)t[id];
    out[id] = (h16)((v - mu) * rsqrtf(var + EPS_) * g[col] + b[col] + emb[col]);
}

// out[e] = dot(h1[e,0:128], w) + bp2  (one wave32 per row)
__global__ __launch_bounds__(256)
void final_dot(const float* __restrict__ h1, const float* __restrict__ w,
               const float* __restrict__ bp2, float* __restrict__ out, int E)
{
    int lane = threadIdx.x & 31;
    int wv   = threadIdx.x >> 5;
    int e    = blockIdx.x * 8 + wv;
    if (e >= E) return;
    const float* row = h1 + (size_t)e * 128;
    float s = 0.f;
#pragma unroll
    for (int k = lane; k < 128; k += 32) s += row[k] * w[k];
    for (int m = 16; m > 0; m >>= 1) s += __shfl_xor(s, m, 32);
    if (lane == 0) out[e] = s + bp2[0];
}

// ---------------------------------------------------------------------------
// Host-side helpers
// ---------------------------------------------------------------------------
static void run_gemm(const h16* A, const h16* Bw, const float* bias, const float* aux,
                     float* C, h16* C16, int M, int N, int K,
                     bool b, bool r, bool x, bool ac, hipStream_t st)
{
    dim3 grid(N / 64, (M + 63) / 64);
    if      ( b && !r && !x && !ac) wmma_gemm<true, false,false,false><<<grid,256,0,st>>>(A,Bw,bias,aux,C,C16,M,N,K);
    else if (!b && !r && !x && !ac) wmma_gemm<false,false,false,false><<<grid,256,0,st>>>(A,Bw,bias,aux,C,C16,M,N,K);
    else if ( b &&  r &&  x && !ac) wmma_gemm<true, true, true, false><<<grid,256,0,st>>>(A,Bw,bias,aux,C,C16,M,N,K);
    else if ( b && !r &&  x && !ac) wmma_gemm<true, false,true, false><<<grid,256,0,st>>>(A,Bw,bias,aux,C,C16,M,N,K);
    else if ( b &&  r &&  x &&  ac) wmma_gemm<true, true, true, true ><<<grid,256,0,st>>>(A,Bw,bias,aux,C,C16,M,N,K);
    else if ( b && !r &&  x &&  ac) wmma_gemm<true, false,true, true ><<<grid,256,0,st>>>(A,Bw,bias,aux,C,C16,M,N,K);
    else if ( b &&  r && !x && !ac) wmma_gemm<true, true, false,false><<<grid,256,0,st>>>(A,Bw,bias,aux,C,C16,M,N,K);
}

static inline int gblk(size_t n) { return (int)((n + 255) / 256); }

// ---------------------------------------------------------------------------
// Entry point
// ---------------------------------------------------------------------------
extern "C" void kernel_launch(void* const* d_in, const int* in_sizes, int n_in,
                              void* d_out, int out_size, void* d_ws, size_t ws_size,
                              hipStream_t stream)
{
    (void)in_sizes; (void)n_in; (void)out_size; (void)ws_size;

    const float* x_gene = (const float*)d_in[0];
    const float* x_dis  = (const float*)d_in[1];
    const float* Win    = (const float*)d_in[2];
    const float* bin_   = (const float*)d_in[3];
    const float* Wself  = (const float*)d_in[4];
    const float* Wneigh = (const float*)d_in[5];
    const float* bsage  = (const float*)d_in[6];
    const float* ln_g   = (const float*)d_in[7];
    const float* ln_b   = (const float*)d_in[8];
    const float* Wv     = (const float*)d_in[9];
    const float* bv     = (const float*)d_in[10];
    const float* Wo     = (const float*)d_in[11];
    const float* bo     = (const float*)d_in[12];
    const float* Wt     = (const float*)d_in[13];
    const float* bt     = (const float*)d_in[14];
    const float* bn_g   = (const float*)d_in[15];
    const float* bn_b   = (const float*)d_in[16];
    const float* et_emb = (const float*)d_in[17];
    const float* Wp1    = (const float*)d_in[18];
    const float* bp1    = (const float*)d_in[19];
    const float* Wp2    = (const float*)d_in[20];
    const float* bp2    = (const float*)d_in[21];
    const int* e_src[3] = { (const int*)d_in[22], (const int*)d_in[24], (const int*)d_in[26] };
    const int* e_dst[3] = { (const int*)d_in[23], (const int*)d_in[25], (const int*)d_in[27] };
    const int* s_src[2] = { (const int*)d_in[28], (const int*)d_in[30] };
    const int* s_dst[2] = { (const int*)d_in[29], (const int*)d_in[31] };
    float* out = (float*)d_out;

    // ---------------- workspace layout ----------------
    char* ws = (char*)d_ws;
    size_t cur = 0;
    auto alloc = [&](size_t bytes) { size_t o = cur; cur += (bytes + 255) & ~(size_t)255; return o; };

    const size_t NGD = (size_t)NG_ * DD_;
    // persistent
    float* hg   = (float*)(ws + alloc(NGD * 4));
    float* hd   = (float*)(ws + alloc(NGD * 4));
    h16*   hg16 = (h16*)  (ws + alloc(NGD * 2));
    h16*   hd16 = (h16*)  (ws + alloc(NGD * 2));
    h16* win16    = (h16*)(ws + alloc((size_t)2 * 65536 * 2));
    h16* wself16  = (h16*)(ws + alloc((size_t)9 * 65536 * 2));
    h16* wneigh16 = (h16*)(ws + alloc((size_t)9 * 65536 * 2));
    h16* wv16     = (h16*)(ws + alloc((size_t)2 * 65536 * 2));
    h16* wo16     = (h16*)(ws + alloc((size_t)2 * 65536 * 2));
    h16* wt16     = (h16*)(ws + alloc((size_t)512 * 256 * 2));
    h16* wp116    = (h16*)(ws + alloc((size_t)256 * 128 * 2));
    size_t union_base = cur;

    // GNN-phase scratch
    h16*   xtmp16 = (h16*)  (ws + alloc(NGD * 2));
    float* agg    = (float*)(ws + alloc(NGD * 4));
    h16*   agg16  = (h16*)  (ws + alloc(NGD * 2));
    float* Pbuf   = (float*)(ws + alloc(NGD * 4));
    float* newg   = (float*)(ws + alloc(NGD * 4));
    float* newd   = (float*)(ws + alloc(NGD * 4));
    float* cnt    = (float*)(ws + alloc((size_t)NG_ * 4));

    // scoring-phase scratch (union with GNN scratch; used after it)
    cur = union_base;
    h16*   ef16  = (h16*)  (ws + alloc((size_t)EP_ * 512 * 2));
    h16*   t16   = (h16*)  (ws + alloc((size_t)EP_ * 256 * 2));
    h16*   tn16  = (h16*)  (ws + alloc((size_t)EP_ * 256 * 2));
    float* h1f   = (float*)(ws + alloc((size_t)EP_ * 128 * 4));
    float* csum  = (float*)(ws + alloc(256 * 4));
    float* csum2 = (float*)(ws + alloc(256 * 4));

    // ---------------- weight conversion (f32 -> f16, L2 resident) ----------------
    cvt_f32_f16<<<gblk(2*65536), 256, 0, stream>>>(Win,    win16,    2*65536);
    cvt_f32_f16<<<gblk(9*65536), 256, 0, stream>>>(Wself,  wself16,  9*65536);
    cvt_f32_f16<<<gblk(9*65536), 256, 0, stream>>>(Wneigh, wneigh16, 9*65536);
    cvt_f32_f16<<<gblk(2*65536), 256, 0, stream>>>(Wv,     wv16,     2*65536);
    cvt_f32_f16<<<gblk(2*65536), 256, 0, stream>>>(Wo,     wo16,     2*65536);
    cvt_f32_f16<<<gblk(512*256), 256, 0, stream>>>(Wt,     wt16,     512*256);
    cvt_f32_f16<<<gblk(256*128), 256, 0, stream>>>(Wp1,    wp116,    256*128);

    // ---------------- input projections ----------------
    cvt_f32_f16<<<gblk(NGD), 256, 0, stream>>>(x_gene, xtmp16, NGD);
    run_gemm(xtmp16, win16,          bin_,       nullptr, hg, hg16, NG_, DD_, DD_, true, false, false, false, stream);
    cvt_f32_f16<<<gblk(NGD), 256, 0, stream>>>(x_dis, xtmp16, NGD);
    run_gemm(xtmp16, win16 + 65536,  bin_ + DD_, nullptr, hd, hd16, ND_, DD_, DD_, true, false, false, false, stream);

    // ---------------- GNN layers ----------------
    const int scat_grid = (int)(((size_t)EM_ * 64 + 255) / 256);
    for (int i = 0; i < 3; ++i) {
        bool relu = (i < 2);
        struct Conv { const float* hsrc; const h16* hself16; int ndst; float* outp; bool acc; };
        Conv convs[3] = {
            { hg, hd16, ND_, newd, false },  // e0: gene->disease
            { hd, hg16, NG_, newg, false },  // e1: disease->gene
            { hg, hg16, NG_, newg, true  },  // e2: gene->gene (sum into newg)
        };
        for (int e = 0; e < 3; ++e) {
            Conv cv = convs[e];
            size_t nD = (size_t)cv.ndst * DD_;
            fill_f32<<<gblk(nD),      256, 0, stream>>>(agg, 0.f, nD);
            fill_f32<<<gblk(cv.ndst), 256, 0, stream>>>(cnt, 0.f, (size_t)cv.ndst);
            scatter_rows<<<scat_grid, 256, 0, stream>>>(cv.hsrc, e_src[e], e_dst[e], agg, EM_);
            scatter_cnt<<<gblk(EM_), 256, 0, stream>>>(e_dst[e], cnt, EM_);
            div_cvt<<<gblk(nD), 256, 0, stream>>>(agg, cnt, agg16, nD);
            // P = h_dst @ Wself
            run_gemm(cv.hself16, wself16 + (size_t)(i*3+e)*65536, nullptr, nullptr,
                     Pbuf, nullptr, cv.ndst, DD_, DD_, false, false, false, false, stream);
            // out (+)= act(m @ Wneigh + b + P)
            run_gemm(agg16, wneigh16 + (size_t)(i*3+e)*65536, bsage + (size_t)(i*3+e)*DD_,
                     Pbuf, cv.outp, nullptr, cv.ndst, DD_, DD_, true, relu, true, cv.acc, stream);
        }

        if (i == 2) {  // length-1 MHA == out_proj(v_proj(x))
            cvt_f32_f16<<<gblk(NGD), 256, 0, stream>>>(newg, xtmp16, NGD);
            run_gemm(xtmp16, wv16,         bv,       nullptr, nullptr, agg16, NG_, DD_, DD_, true, false, false, false, stream);
            run_gemm(agg16,  wo16,         bo,       nullptr, newg,  nullptr, NG_, DD_, DD_, true, false, false, false, stream);
            cvt_f32_f16<<<gblk(NGD), 256, 0, stream>>>(newd, xtmp16, NGD);
            run_gemm(xtmp16, wv16 + 65536, bv + DD_, nullptr, nullptr, agg16, ND_, DD_, DD_, true, false, false, false, stream);
            run_gemm(agg16,  wo16 + 65536, bo + DD_, nullptr, newd,  nullptr, ND_, DD_, DD_, true, false, false, false, stream);
        }

        const float* res_g = (i > 0) ? hg : nullptr;
        const float* res_d = (i > 0) ? hd : nullptr;
        ln_res<<<NG_, 256, 0, stream>>>(newg, res_g, ln_g + (size_t)(i*2+0)*DD_, ln_b + (size_t)(i*2+0)*DD_, hg, hg16);
        ln_res<<<ND_, 256, 0, stream>>>(newd, res_d, ln_g + (size_t)(i*2+1)*DD_, ln_b + (size_t)(i*2+1)*DD_, hd, hd16);
    }

    // ---------------- link scoring (pos, neg) ----------------
    for (int s = 0; s < 2; ++s) {
        gather_concat<<<gblk((size_t)EP_*128), 256, 0, stream>>>(hg16, hd16, s_src[s], s_dst[s], ef16, EP_);
        // t = relu(ef @ Wt + bt)  -> f16
        run_gemm(ef16, wt16, bt, nullptr, nullptr, t16, EP_, 256, 512, true, true, false, false, stream);
        // BatchNorm (batch stats) + et_emb
        fill_f32<<<1, 256, 0, stream>>>(csum,  0.f, 256);
        fill_f32<<<1, 256, 0, stream>>>(csum2, 0.f, 256);
        bn_stats<<<EP_/256, 256, 0, stream>>>(t16, csum, csum2, 256);
        bn_apply<<<gblk((size_t)EP_*256), 256, 0, stream>>>(t16, csum, csum2, bn_g, bn_b, et_emb,
                                                            tn16, (size_t)EP_*256, 1.f/(float)EP_);
        // h1 = relu(tn @ Wp1 + bp1)
        run_gemm(tn16, wp116, bp1, nullptr, h1f, nullptr, EP_, 128, 256, true, true, false, false, stream);
        // out = h1 @ Wp2 + bp2
        final_dot<<<EP_/8, 256, 0, stream>>>(h1f, Wp2, bp2, out + (size_t)s * EP_, EP_);
    }
}
